// NetGINEAlchemy_6828998001136
// MI455X (gfx1250) — matrix operand order repaired and checked by reading, not compiled, not run on hardware
//
#include <hip/hip_runtime.h>
#include <hip/hip_bf16.h>
#include <math.h>

typedef _Float16 v16h __attribute__((ext_vector_type(16)));
typedef _Float16 v8h  __attribute__((ext_vector_type(8)));
typedef float    v8f  __attribute__((ext_vector_type(8)));

__device__ __forceinline__ float sigmoidf_(float x) { return 1.0f / (1.0f + __expf(-x)); }
// order-preserving float <-> uint for atomicMax on floats (handles negatives)
__device__ __forceinline__ unsigned f2o(float f) {
    unsigned u = __float_as_uint(f);
    return (u & 0x80000000u) ? ~u : (u | 0x80000000u);
}
__device__ __forceinline__ float o2f(unsigned m) {
    unsigned u = (m & 0x80000000u) ? (m & 0x7FFFFFFFu) : ~m;
    return __uint_as_float(u);
}
#define EMAX_NEG_INF 0x007FFFFFu  // f2o(-inf)

// ---------------------------------------------------------------- utilities
__global__ void zero_f32(float* __restrict__ p, long n) {
    long i = (long)blockIdx.x * blockDim.x + threadIdx.x;
    if (i < n) p[i] = 0.0f;
}

// Pre-swizzle a [K x 64] f32 weight matrix into WMMA B-fragment order (f16):
// frag[((kt*4 + j)*32 + lane)*16 + i] = w[(kt*32 + (lane>>4)*8 + (i&7) + (i>>3)*16)*64 + j*16 + (lane&15)]
// so a wave can fetch one whole fragment as a single 32-byte vector load.
__global__ void pack_bfrag(const float* __restrict__ w, _Float16* __restrict__ frag, int KT) {
    int idx = blockIdx.x * blockDim.x + threadIdx.x;
    if (idx >= KT * 2048) return;
    int i    = idx & 15;
    int lane = (idx >> 4) & 31;
    int j    = (idx >> 9) & 3;
    int kt   = idx >> 11;
    int n = j * 16 + (lane & 15);
    int k = kt * 32 + (lane >> 4) * 8 + (i & 7) + (i >> 3) * 16;
    frag[idx] = (_Float16)w[k * 64 + n];
}

// Transpose [16 x DL] bond-encoder weight to [DL x 16] for contiguous per-channel reads
__global__ void pack_beWT(const float* __restrict__ w, float* __restrict__ wt, int DL) {
    int idx = blockIdx.x * blockDim.x + threadIdx.x;
    if (idx >= DL * 16) return;
    int k = idx & 15, d = idx >> 4;
    wt[d * 16 + k] = w[k * DL + d];
}

// --------------------------------------------------- edge message + scatter
// m = relu(x[src] + edge_attr@beW + beB) * ew ; atomic segment-sum at dst
template <int DL>
__global__ __launch_bounds__(256) void edge_msg(
    const float* __restrict__ ea, const float* __restrict__ ew,
    const int* __restrict__ src, const int* __restrict__ dst,
    const float* __restrict__ beWT,  // [DL x 16] transposed
    const float* __restrict__ beB,
    const float* __restrict__ x, float* __restrict__ agg, int E)
{
    constexpr int LOG = (DL == 32) ? 5 : 6;
    long idx = (long)blockIdx.x * blockDim.x + threadIdx.x;
    int e = (int)(idx >> LOG);
    int d = (int)(idx & (DL - 1));
    if (e >= E) return;
    __builtin_prefetch(ea + (long)e * 16 + 1024, 0, 1);  // global_prefetch_b8
    const float4* eav = (const float4*)(ea + (long)e * 16);
    const float4* wtv = (const float4*)(beWT + d * 16);
    float4 e0 = eav[0], e1 = eav[1], e2 = eav[2], e3 = eav[3];
    float4 w0 = wtv[0], w1 = wtv[1], w2 = wtv[2], w3 = wtv[3];
    float acc = beB[d];
    acc += e0.x * w0.x + e0.y * w0.y + e0.z * w0.z + e0.w * w0.w;
    acc += e1.x * w1.x + e1.y * w1.y + e1.z * w1.z + e1.w * w1.w;
    acc += e2.x * w2.x + e2.y * w2.y + e2.z * w2.z + e2.w * w2.w;
    acc += e3.x * w3.x + e3.y * w3.y + e3.z * w3.z + e3.w * w3.w;
    float m = x[(long)src[e] * DL + d] + acc;
    m = fmaxf(m, 0.0f) * ew[e];
    atomicAdd(&agg[(long)dst[e] * DL + d], m);
}

// ----------------------------------------------- fused node MLP via WMMA
// z = (1+eps)*x + agg ; x_out = relu( relu(z@w1+b1) @ w2 + b2 )
// one wave = one 16-row tile; f16 operands, f32 accumulate; weights pre-swizzled
template <int FIN>
__global__ __launch_bounds__(256) void node_mlp_wmma(
    const float* __restrict__ xin, const float* __restrict__ agg,
    const _Float16* __restrict__ w1f,  // fragment-packed, KT1*4 fragments
    const float* __restrict__ b1,
    const _Float16* __restrict__ w2f,  // fragment-packed, 2*4 fragments
    const float* __restrict__ b2,
    const float* __restrict__ epsp, float* __restrict__ xout, int nNodes)
{
    constexpr int KT1 = FIN / 32;
    __shared__ _Float16 htile[8][16 * 64];  // 16 KB
    const int wave = threadIdx.x >> 5;
    const int lane = threadIdx.x & 31;
    const int mloc = lane & 15;
    const int half = lane >> 4;
    const int m0 = (blockIdx.x * 8 + wave) * 16;
    const float eps1 = 1.0f + epsp[0];
    int row = m0 + mloc;
    int rowc = (row < nNodes) ? row : (nNodes - 1);  // branchless tail clamp

    // A1 fragments: per lane two contiguous 8-float runs per k-tile -> float4 loads
    const float* xrow = xin + (long)rowc * FIN;
    const float* arow = agg + (long)rowc * FIN;
    v16h a1[KT1];
#pragma unroll
    for (int kt = 0; kt < KT1; ++kt) {
        int k0 = kt * 32 + half * 8;        // elements i=0..7 -> K = k0+i
        int k1 = k0 + 16;                   // elements i=8..15 -> K = k1+(i-8)
        float4 xa = *(const float4*)(xrow + k0);
        float4 xb = *(const float4*)(xrow + k0 + 4);
        float4 xc = *(const float4*)(xrow + k1);
        float4 xd = *(const float4*)(xrow + k1 + 4);
        float4 ga = *(const float4*)(arow + k0);
        float4 gb = *(const float4*)(arow + k0 + 4);
        float4 gc = *(const float4*)(arow + k1);
        float4 gd = *(const float4*)(arow + k1 + 4);
        a1[kt][0]  = (_Float16)(eps1 * xa.x + ga.x);
        a1[kt][1]  = (_Float16)(eps1 * xa.y + ga.y);
        a1[kt][2]  = (_Float16)(eps1 * xa.z + ga.z);
        a1[kt][3]  = (_Float16)(eps1 * xa.w + ga.w);
        a1[kt][4]  = (_Float16)(eps1 * xb.x + gb.x);
        a1[kt][5]  = (_Float16)(eps1 * xb.y + gb.y);
        a1[kt][6]  = (_Float16)(eps1 * xb.z + gb.z);
        a1[kt][7]  = (_Float16)(eps1 * xb.w + gb.w);
        a1[kt][8]  = (_Float16)(eps1 * xc.x + gc.x);
        a1[kt][9]  = (_Float16)(eps1 * xc.y + gc.y);
        a1[kt][10] = (_Float16)(eps1 * xc.z + gc.z);
        a1[kt][11] = (_Float16)(eps1 * xc.w + gc.w);
        a1[kt][12] = (_Float16)(eps1 * xd.x + gd.x);
        a1[kt][13] = (_Float16)(eps1 * xd.y + gd.y);
        a1[kt][14] = (_Float16)(eps1 * xd.z + gd.z);
        a1[kt][15] = (_Float16)(eps1 * xd.w + gd.w);
    }

    // GEMM1: fragment B loads are single 32B vector loads from packed weights
#pragma unroll
    for (int j = 0; j < 4; ++j) {
        v8f acc = {};
        int n = j * 16 + mloc;
#pragma unroll
        for (int kt = 0; kt < KT1; ++kt) {
            v16h b = *(const v16h*)(w1f + ((kt * 4 + j) * 32 + lane) * 16);
            acc = __builtin_amdgcn_wmma_f32_16x16x32_f16(false, a1[kt], false, b,
                                                         (short)0, acc, false, false);
        }
        float bias = b1[n];
#pragma unroll
        for (int r = 0; r < 8; ++r)
            htile[wave][(r + 8 * half) * 64 + n] = (_Float16)fmaxf(acc[r] + bias, 0.0f);
    }
    __syncthreads();

    // A2 fragments from f16 LDS tile: two 16B ds reads per k-tile
    v16h a2[2];
#pragma unroll
    for (int kt = 0; kt < 2; ++kt) {
        int k0 = kt * 32 + half * 8;
        v8h lo = *(const v8h*)&htile[wave][mloc * 64 + k0];
        v8h hi = *(const v8h*)&htile[wave][mloc * 64 + k0 + 16];
#pragma unroll
        for (int i = 0; i < 8; ++i) { a2[kt][i] = lo[i]; a2[kt][8 + i] = hi[i]; }
    }

    // GEMM2 + bias + relu -> global
#pragma unroll
    for (int j = 0; j < 4; ++j) {
        v8f acc = {};
        int n = j * 16 + mloc;
#pragma unroll
        for (int kt = 0; kt < 2; ++kt) {
            v16h b = *(const v16h*)(w2f + ((kt * 4 + j) * 32 + lane) * 16);
            acc = __builtin_amdgcn_wmma_f32_16x16x32_f16(false, a2[kt], false, b,
                                                         (short)0, acc, false, false);
        }
        float bias = b2[n];
#pragma unroll
        for (int r = 0; r < 8; ++r) {
            int grow = m0 + r + 8 * half;
            if (grow < nNodes) xout[(long)grow * 64 + n] = fmaxf(acc[r] + bias, 0.0f);
        }
    }
}

// ------------------------------------------------------ Set2Set LSTM step
__global__ __launch_bounds__(256) void lstm_step(
    const float* __restrict__ wih, const float* __restrict__ whh,
    const float* __restrict__ bih, const float* __restrict__ bhh,
    float* __restrict__ qstar, float* __restrict__ hG, float* __restrict__ cG,
    float* __restrict__ rG, float* __restrict__ asum, unsigned* __restrict__ emaxU)
{
    int g = blockIdx.x;
    int t = threadIdx.x;  // 256 = 4*D gate outputs
    __shared__ float qs[128], hs[64], gts[256];
    if (t < 128) qs[t] = qstar[g * 128 + t];
    if (t < 64) hs[t] = hG[g * 64 + t];
    __syncthreads();
    float acc = bih[t] + bhh[t];
#pragma unroll 8
    for (int k = 0; k < 128; ++k) acc += qs[k] * wih[t * 128 + k];
#pragma unroll 8
    for (int k = 0; k < 64; ++k) acc += hs[k] * whh[t * 64 + k];
    gts[t] = acc;
    __syncthreads();
    if (t < 64) {
        float ig = sigmoidf_(gts[t]);
        float fg = sigmoidf_(gts[64 + t]);
        float gg = tanhf(gts[128 + t]);
        float og = sigmoidf_(gts[192 + t]);
        float c = fg * cG[g * 64 + t] + ig * gg;
        cG[g * 64 + t] = c;
        float h = og * tanhf(c);
        hG[g * 64 + t] = h;
        qstar[g * 128 + t] = h;  // q part of q_star
        rG[g * 64 + t] = 0.0f;   // reset attention readout
    }
    if (t == 64) asum[g] = 0.0f;
    if (t == 65) emaxU[g] = EMAX_NEG_INF;
}

// e[n] = <x[n], q[batch[n]]>; segment max via ordered-uint atomicMax
__global__ __launch_bounds__(256) void attn_e(
    const float* __restrict__ x, const float* __restrict__ hG,
    const int* __restrict__ batch, float* __restrict__ eN,
    unsigned* __restrict__ emaxU, int N)
{
    int n = blockIdx.x * blockDim.x + threadIdx.x;
    if (n >= N) return;
    int g = batch[n];
    float dot = 0.0f;
    const float4* xv = (const float4*)(x + (long)n * 64);
    const float4* hv = (const float4*)(hG + (long)g * 64);
#pragma unroll
    for (int d = 0; d < 16; ++d) {
        float4 a = xv[d], b = hv[d];
        dot += a.x * b.x + a.y * b.y + a.z * b.z + a.w * b.w;
    }
    eN[n] = dot;
    atomicMax(&emaxU[g], f2o(dot));
}

// a = exp(e - emax); segment sum
__global__ __launch_bounds__(256) void attn_a(
    const int* __restrict__ batch, float* __restrict__ eN,
    const unsigned* __restrict__ emaxU, float* __restrict__ asum, int N)
{
    int n = blockIdx.x * blockDim.x + threadIdx.x;
    if (n >= N) return;
    int g = batch[n];
    float emax = o2f(emaxU[g]);
    if (!isfinite(emax)) emax = 0.0f;
    float a = __expf(eN[n] - emax);
    eN[n] = a;
    atomicAdd(&asum[g], a);
}

// r[g] += (a/asum) * x[n]   (thread per (node, channel))
__global__ __launch_bounds__(256) void attn_r(
    const float* __restrict__ x, const int* __restrict__ batch,
    const float* __restrict__ eN, const float* __restrict__ asum,
    float* __restrict__ rG, int N)
{
    long idx = (long)blockIdx.x * blockDim.x + threadIdx.x;
    int n = (int)(idx >> 6);
    int d = (int)(idx & 63);
    if (n >= N) return;
    int g = batch[n];
    float coef = eN[n] / (asum[g] + 1e-16f);
    atomicAdd(&rG[g * 64 + d], coef * x[(long)n * 64 + d]);
}

__global__ void copy_r_to_qstar(const float* __restrict__ rG, float* __restrict__ qstar, int G) {
    int i = blockIdx.x * blockDim.x + threadIdx.x;
    if (i >= G * 64) return;
    int g = i >> 6, d = i & 63;
    qstar[g * 128 + 64 + d] = rG[i];
}

// -------------------------------------------------------------- MLP head
__global__ __launch_bounds__(64) void head_mlp(
    const float* __restrict__ qstar, const float* __restrict__ fc1w,
    const float* __restrict__ fc1b, const float* __restrict__ fc4w,
    const float* __restrict__ fc4b, float* __restrict__ out)
{
    int g = blockIdx.x, d = threadIdx.x;  // 64 threads
    __shared__ float qs[128], h1[64];
    qs[d] = qstar[g * 128 + d];
    qs[64 + d] = qstar[g * 128 + 64 + d];
    __syncthreads();
    float acc = fc1b[d];
#pragma unroll 8
    for (int k = 0; k < 128; ++k) acc += qs[k] * fc1w[k * 64 + d];
    h1[d] = fmaxf(acc, 0.0f);
    __syncthreads();
    if (d < 12) {
        float o = fc4b[d];
#pragma unroll 8
        for (int k = 0; k < 64; ++k) o += h1[k] * fc4w[k * 12 + d];
        out[g * 12 + d] = o;
    }
}

// ---------------------------------------------------------------- driver
extern "C" void kernel_launch(void* const* d_in, const int* in_sizes, int n_in,
                              void* d_out, int out_size, void* d_ws, size_t ws_size,
                              hipStream_t stream) {
    const float* x0        = (const float*)d_in[0];
    const float* edge_attr = (const float*)d_in[1];
    const float* ew        = (const float*)d_in[2];
    const int*   eidx      = (const int*)d_in[3];
    const int*   batch     = (const int*)d_in[4];
    const float* c0_beW = (const float*)d_in[5];
    const float* c0_beB = (const float*)d_in[6];
    const float* c0_w1  = (const float*)d_in[7];
    const float* c0_b1  = (const float*)d_in[8];
    const float* c0_w2  = (const float*)d_in[9];
    const float* c0_b2  = (const float*)d_in[10];
    const float* c0_eps = (const float*)d_in[11];
    const float* cs_beW = (const float*)d_in[12];
    const float* cs_beB = (const float*)d_in[13];
    const float* cs_w1  = (const float*)d_in[14];
    const float* cs_b1  = (const float*)d_in[15];
    const float* cs_w2  = (const float*)d_in[16];
    const float* cs_b2  = (const float*)d_in[17];
    const float* cs_eps = (const float*)d_in[18];
    const float* lstm_wih = (const float*)d_in[19];
    const float* lstm_whh = (const float*)d_in[20];
    const float* lstm_bih = (const float*)d_in[21];
    const float* lstm_bhh = (const float*)d_in[22];
    const float* fc1_w = (const float*)d_in[23];
    const float* fc1_b = (const float*)d_in[24];
    const float* fc4_w = (const float*)d_in[25];
    const float* fc4_b = (const float*)d_in[26];

    const int N = in_sizes[0] / 32;   // 100000
    const int E = in_sizes[1] / 16;   // 1000000
    const int G = out_size / 12;      // 2000
    const int* srcI = eidx;
    const int* dstI = eidx + E;

    // workspace carving (256B aligned)
    char* w = (char*)d_ws;
    auto carve = [&](size_t bytes) {
        void* p = (void*)w;
        w += (bytes + 255) & ~(size_t)255;
        return p;
    };
    float*     xws   = (float*)carve(sizeof(float) * (size_t)N * 64);
    float*     agg   = (float*)carve(sizeof(float) * (size_t)N * 64);
    _Float16*  w1f   = (_Float16*)carve(sizeof(_Float16) * (2048 + 5 * 4096));  // frag-packed w1
    _Float16*  w2f   = (_Float16*)carve(sizeof(_Float16) * (6 * 4096));         // frag-packed w2
    float*     beT0  = (float*)carve(sizeof(float) * 32 * 16);                  // transposed beW (layer0)
    float*     beT   = (float*)carve(sizeof(float) * 5 * 64 * 16);              // transposed beW (layers 1-5)
    float*     s2s   = (float*)carve(sizeof(float) * (size_t)G * 256);          // qstar|h|c
    float*     qstar = s2s;
    float*     hG    = s2s + (size_t)G * 128;
    float*     cG    = s2s + (size_t)G * 192;
    float*     eN    = (float*)carve(sizeof(float) * (size_t)N);
    unsigned*  emaxU = (unsigned*)carve(sizeof(unsigned) * (size_t)G);
    float*     asum  = (float*)carve(sizeof(float) * (size_t)G);
    float*     rG    = (float*)carve(sizeof(float) * (size_t)G * 64);
    (void)ws_size; (void)n_in;

    // weight staging: WMMA fragment packing + bond-encoder transpose
    pack_bfrag<<<(2048 + 255) / 256, 256, 0, stream>>>(c0_w1, w1f, 1);
    pack_bfrag<<<(4096 + 255) / 256, 256, 0, stream>>>(c0_w2, w2f, 2);
    for (int l = 0; l < 5; ++l) {
        pack_bfrag<<<(4096 + 255) / 256, 256, 0, stream>>>(
            cs_w1 + (size_t)l * 64 * 64, w1f + 2048 + (size_t)l * 4096, 2);
        pack_bfrag<<<(4096 + 255) / 256, 256, 0, stream>>>(
            cs_w2 + (size_t)l * 64 * 64, w2f + 4096 + (size_t)l * 4096, 2);
    }
    pack_beWT<<<(32 * 16 + 255) / 256, 256, 0, stream>>>(c0_beW, beT0, 32);
    for (int l = 0; l < 5; ++l)
        pack_beWT<<<(64 * 16 + 255) / 256, 256, 0, stream>>>(
            cs_beW + (size_t)l * 16 * 64, beT + (size_t)l * 64 * 16, 64);

    const int tiles = (N + 15) / 16;
    const int mlpBlocks = (tiles + 7) / 8;

    // layer 0 : Fin=32 -> D=64
    zero_f32<<<((long)N * 32 + 255) / 256, 256, 0, stream>>>(agg, (long)N * 32);
    edge_msg<32><<<(unsigned)(((long)E * 32 + 255) / 256), 256, 0, stream>>>(
        edge_attr, ew, srcI, dstI, beT0, c0_beB, x0, agg, E);
    node_mlp_wmma<32><<<mlpBlocks, 256, 0, stream>>>(
        x0, agg, w1f, c0_b1, w2f, c0_b2, c0_eps, xws, N);

    // layers 1..5 : D=64 -> D=64 (in-place on xws)
    for (int l = 0; l < 5; ++l) {
        zero_f32<<<((long)N * 64 + 255) / 256, 256, 0, stream>>>(agg, (long)N * 64);
        edge_msg<64><<<(unsigned)(((long)E * 64 + 255) / 256), 256, 0, stream>>>(
            edge_attr, ew, srcI, dstI, beT + (size_t)l * 64 * 16, cs_beB + l * 64,
            xws, agg, E);
        node_mlp_wmma<64><<<mlpBlocks, 256, 0, stream>>>(
            xws, agg, w1f + 2048 + (size_t)l * 4096, cs_b1 + l * 64,
            w2f + 4096 + (size_t)l * 4096, cs_b2 + l * 64, cs_eps + l, xws, N);
    }

    // Set2Set
    zero_f32<<<((long)G * 256 + 255) / 256, 256, 0, stream>>>(s2s, (long)G * 256);
    for (int s = 0; s < 6; ++s) {
        lstm_step<<<G, 256, 0, stream>>>(lstm_wih, lstm_whh, lstm_bih, lstm_bhh,
                                         qstar, hG, cG, rG, asum, emaxU);
        attn_e<<<(N + 255) / 256, 256, 0, stream>>>(xws, hG, batch, eN, emaxU, N);
        attn_a<<<(N + 255) / 256, 256, 0, stream>>>(batch, eN, emaxU, asum, N);
        attn_r<<<(unsigned)(((long)N * 64 + 255) / 256), 256, 0, stream>>>(
            xws, batch, eN, asum, rG, N);
        copy_r_to_qstar<<<(G * 64 + 255) / 256, 256, 0, stream>>>(rG, qstar, G);
    }

    head_mlp<<<G, 64, 0, stream>>>(qstar, fc1_w, fc1_b, fc4_w, fc4_b, (float*)d_out);
}
